// SwinTransformerBlock_82154134438466
// MI455X (gfx1250) — compile-verified, matrix operand order checked by reading, not run on hardware
//
#include <hip/hip_runtime.h>

// ---------------------------------------------------------------------------
// Types
// ---------------------------------------------------------------------------
typedef __bf16 bf16;
typedef bf16  v16bf __attribute__((ext_vector_type(16)));
typedef bf16  v8bf  __attribute__((ext_vector_type(8)));
typedef float v8f   __attribute__((ext_vector_type(8)));

#define B_   8
#define L_   4096
#define C_   512
#define WS_  64
#define SHIFT_ 32
#define H_   16
#define HD_  32
#define HID_ 2048
#define NW_  (L_ / WS_)      // 64
#define MT_  (B_ * L_)       // 32768 token rows

__device__ __forceinline__ bf16 f2bf(float f) {
    union { float f; unsigned u; } cv; cv.f = f;
    unsigned u = cv.u;
    u += 0x7FFFu + ((u >> 16) & 1u);          // round-to-nearest-even
    unsigned short s = (unsigned short)(u >> 16);
    union { unsigned short s; bf16 b; } ov; ov.s = s;
    return ov.b;
}

// --- CDNA5 async global->LDS copy (16B per lane), tracked by ASYNCcnt ------
__device__ __forceinline__ void async_ld16(const void* g, void* l) {
    unsigned lofs = (unsigned)(size_t)(__attribute__((address_space(3))) char*)l;
    asm volatile("global_load_async_to_lds_b128 %0, %1, off"
                 :: "v"(lofs), "v"((unsigned long long)(size_t)g)
                 : "memory");
}

template <int N>
__device__ __forceinline__ void wait_async() {
#if __has_builtin(__builtin_amdgcn_s_wait_asynccnt)
    __builtin_amdgcn_s_wait_asynccnt((unsigned short)N);
#else
    asm volatile("s_wait_asynccnt %0" :: "n"(N) : "memory");
#endif
}

// Load one 16x32 bf16 WMMA A/B fragment from an LDS tile (row-major, stride
// `ld` elements). Per CDNA5 ISA 16-bit A layout: lane<16 -> row=lane, K in
// {kb..kb+7, kb+16..kb+23} with kb=0; lane>=16 -> row=lane-16, kb=8.
// B fragments use the same pattern on an N x K (transposed) tile.
__device__ __forceinline__ v16bf load_frag(const bf16* base, int row0, int ld) {
    int lane = threadIdx.x & 31;
    const bf16* p = base + (size_t)(row0 + (lane & 15)) * ld + ((lane >> 4) << 3);
    v8bf lo = *(const v8bf*)p;         // K = kb .. kb+7
    v8bf hi = *(const v8bf*)(p + 16);  // K = kb+16 .. kb+23
    return __builtin_shufflevector(lo, hi, 0,1,2,3,4,5,6,7,8,9,10,11,12,13,14,15);
}

__device__ __forceinline__ v8f wmma_bf16(v16bf a, v16bf b, v8f c) {
    return __builtin_amdgcn_wmma_f32_16x16x32_bf16(false, a, false, b, (short)0, c,
                                                   false, false);
}

// ---------------------------------------------------------------------------
// Elementwise prep kernels
// ---------------------------------------------------------------------------
__global__ void cvt_kernel(const float* __restrict__ s, bf16* __restrict__ d, long n) {
    long i  = (long)blockIdx.x * blockDim.x + threadIdx.x;
    long st = (long)gridDim.x * blockDim.x;
    for (; i < n; i += st) d[i] = f2bf(s[i]);
}

// xw[b*L + l][c] = x[b][(l+SHIFT) mod L][c], converted to bf16
__global__ void shiftcvt_kernel(const float* __restrict__ x, bf16* __restrict__ d) {
    long n  = (long)MT_ * C_;
    long i  = (long)blockIdx.x * blockDim.x + threadIdx.x;
    long st = (long)gridDim.x * blockDim.x;
    for (; i < n; i += st) {
        long m = i >> 9;                 // / C_
        int  c = (int)(i & (C_ - 1));
        int  b = (int)(m >> 12);         // / L_
        int  l = (int)(m & (L_ - 1));
        int  ls = (l + SHIFT_) & (L_ - 1);
        d[i] = f2bf(x[((size_t)b * L_ + ls) * C_ + c]);
    }
}

__global__ void qkvb_kernel(const float* __restrict__ qb, const float* __restrict__ vb,
                            float* __restrict__ qkvb) {
    int i = blockIdx.x * blockDim.x + threadIdx.x;
    if (i < 3 * C_)
        qkvb[i] = (i < C_) ? qb[i] : ((i < 2 * C_) ? 0.0f : vb[i - 2 * C_]);
}

// Continuous position bias MLP: tbl[r][h], r in 0..126, h in 0..15
__global__ __launch_bounds__(128) void cpb_kernel(const float* __restrict__ w1,
                                                  const float* __restrict__ b1,
                                                  const float* __restrict__ w2,
                                                  float* __restrict__ tbl) {
    __shared__ float hid[512];
    int r = blockIdx.x;                          // 0..126
    float rel = (float)(r - (WS_ - 1));
    float xc  = rel * (8.0f / (float)(WS_ - 1));
    float sgn = (xc > 0.f) ? 1.f : ((xc < 0.f) ? -1.f : 0.f);
    float t   = sgn * log2f(fabsf(xc) + 1.0f) * (1.0f / 3.0f);   // /log2(8)
    for (int j = threadIdx.x; j < 512; j += 128)
        hid[j] = fmaxf(t * w1[j] + b1[j], 0.0f);
    __syncthreads();
    if (threadIdx.x < H_) {
        int hh = threadIdx.x;
        float s = 0.f;
        for (int j = 0; j < 512; ++j) s += hid[j] * w2[hh * 512 + j];
        tbl[r * H_ + hh] = s;
    }
}

// ---------------------------------------------------------------------------
// Tiled bf16 GEMM:  out[M,N] = A[M,K] @ B[N,K]^T (+bias)  via v_wmma bf16.
// BM=BN=128, BK=32, 256 threads = 8 waves (4Mx2N), each wave owns a 32x64
// sub-tile = 2x4 WMMA accumulators. Tiles are staged with CDNA5 async
// global->LDS DMA (ASYNCcnt) into double-buffered LDS: while buffer `cur`
// feeds the WMMAs, buffer `cur^1` is being filled by the DMA engine.
// EPI 0: fp32 store with bias.  EPI 1: bf16 store with bias + exact GELU.
// ---------------------------------------------------------------------------
template <int EPI>
__global__ __launch_bounds__(256)
void gemm_bf16(const bf16* __restrict__ A, const bf16* __restrict__ Bm,
               const float* __restrict__ bias,
               float* __restrict__ outF, bf16* __restrict__ outB,
               int M, int N, int K) {
    __shared__ bf16 As[2][128][40];
    __shared__ bf16 Bs[2][128][40];

    const int tid = threadIdx.x;
    const int m0  = blockIdx.y * 128;
    const int n0  = blockIdx.x * 128;

    const int lr = tid >> 1;        // 0..127 : tile row loaded by this thread
    const int lh = tid & 1;         // half of the 32-wide K slab

    const int wid = tid >> 5;
    const int wm  = wid >> 1;       // 0..3
    const int wn  = wid & 1;        // 0..1
    const int lane  = tid & 31;
    const int ncol  = lane & 15;
    const int mbase = (lane >> 4) << 3;

    v8f acc[2][4];
#pragma unroll
    for (int i = 0; i < 2; ++i)
#pragma unroll
        for (int j = 0; j < 4; ++j)
#pragma unroll
            for (int r = 0; r < 8; ++r) acc[i][j][r] = 0.f;

    const int nk = K >> 5;

    // 4 async 16B DMAs per thread per stage (2 for A, 2 for B)
    auto issue = [&](int ks, int buf) {
        const bf16* pa = A  + (size_t)(m0 + lr) * K + ks * 32 + lh * 16;
        const bf16* pb = Bm + (size_t)(n0 + lr) * K + ks * 32 + lh * 16;
        async_ld16(pa,     &As[buf][lr][lh * 16]);
        async_ld16(pa + 8, &As[buf][lr][lh * 16 + 8]);
        async_ld16(pb,     &Bs[buf][lr][lh * 16]);
        async_ld16(pb + 8, &Bs[buf][lr][lh * 16 + 8]);
    };

    issue(0, 0);
    for (int ks = 0; ks < nk; ++ks) {
        const int cur = ks & 1;
        if (ks + 1 < nk) {
            issue(ks + 1, cur ^ 1);   // prefetch next stage into other buffer
            wait_async<4>();          // oldest 4 (stage ks) have landed in LDS
        } else {
            wait_async<0>();
        }
        __syncthreads();              // all waves' DMAs for stage ks visible

        v16bf af[2], bfr[4];
#pragma unroll
        for (int i = 0; i < 2; ++i)
            af[i] = load_frag(&As[cur][0][0], 32 * wm + 16 * i, 40);
#pragma unroll
        for (int j = 0; j < 4; ++j)
            bfr[j] = load_frag(&Bs[cur][0][0], 64 * wn + 16 * j, 40);
#pragma unroll
        for (int i = 0; i < 2; ++i)
#pragma unroll
            for (int j = 0; j < 4; ++j) acc[i][j] = wmma_bf16(af[i], bfr[j], acc[i][j]);

        __syncthreads();              // done reading `cur` before it is refilled
    }

#pragma unroll
    for (int i = 0; i < 2; ++i)
#pragma unroll
        for (int j = 0; j < 4; ++j) {
            int gm = m0 + 32 * wm + 16 * i + mbase;
            int gn = n0 + 64 * wn + 16 * j + ncol;
            float bv = bias ? bias[gn] : 0.f;
#pragma unroll
            for (int r = 0; r < 8; ++r) {
                float v = acc[i][j][r] + bv;
                if (EPI == 1) v = 0.5f * v * (1.0f + erff(v * 0.70710678f));
                if (EPI == 0) outF[(size_t)(gm + r) * N + gn] = v;
                else          outB[(size_t)(gm + r) * N + gn] = f2bf(v);
            }
        }
}

// ---------------------------------------------------------------------------
// Attention: one block (64 threads / 2 waves) per (window, head).
// Cosine attention with per-head clamped temperature folded into q, CPB bias,
// shift mask on the last window, softmax, then P @ V — all matmuls via WMMA.
// ---------------------------------------------------------------------------
__global__ __launch_bounds__(64)
void attn_kernel(const float* __restrict__ qkv, const float* __restrict__ logit_scale,
                 const float* __restrict__ tbl, bf16* __restrict__ attn_out) {
    __shared__ bf16  Aq[64][40];
    __shared__ bf16  Ak[64][40];
    __shared__ bf16  Vt[32][72];     // v transposed: [d][k]
    __shared__ float Sm[64][65];
    __shared__ bf16  Pm[64][72];

    const int win  = blockIdx.x >> 4;
    const int head = blockIdx.x & (H_ - 1);
    const int t    = threadIdx.x;                 // 0..63 == token row in window
    const bool masked = ((win & (NW_ - 1)) == NW_ - 1);

    const float sc = __expf(fminf(logit_scale[head], 4.6051702f)); // ln(100)

    // ---- load + normalize this row of q/k, stage v transposed ----
    const float* row = qkv + (size_t)(win * WS_ + t) * (3 * C_) + head * HD_;
    float qv[HD_], kv[HD_], qs = 0.f, ks = 0.f;
#pragma unroll
    for (int d = 0; d < HD_; ++d) {
        qv[d] = row[d];          qs += qv[d] * qv[d];
        kv[d] = row[C_ + d];     ks += kv[d] * kv[d];
    }
    float qn = sc / fmaxf(sqrtf(qs), 1e-12f);
    float kn = 1.0f / fmaxf(sqrtf(ks), 1e-12f);
#pragma unroll
    for (int d = 0; d < HD_; ++d) {
        Aq[t][d] = f2bf(qv[d] * qn);
        Ak[t][d] = f2bf(kv[d] * kn);
        Vt[d][t] = f2bf(row[2 * C_ + d]);
    }
    __syncthreads();

    const int w     = t >> 5;          // wave id: rows 32w..32w+31 of S
    const int lane  = t & 31;
    const int ncol  = lane & 15;
    const int mbase = (lane >> 4) << 3;

    // ---- S = (scaled qn) @ kn^T : 2x4 tiles, K=32 (one WMMA each) ----
    {
        v8f acc[2][4];
#pragma unroll
        for (int i = 0; i < 2; ++i)
#pragma unroll
            for (int j = 0; j < 4; ++j)
#pragma unroll
                for (int r = 0; r < 8; ++r) acc[i][j][r] = 0.f;

        v16bf af[2], bfr[4];
#pragma unroll
        for (int i = 0; i < 2; ++i) af[i] = load_frag(&Aq[0][0], 32 * w + 16 * i, 40);
#pragma unroll
        for (int j = 0; j < 4; ++j) bfr[j] = load_frag(&Ak[0][0], 16 * j, 40);
#pragma unroll
        for (int i = 0; i < 2; ++i)
#pragma unroll
            for (int j = 0; j < 4; ++j) acc[i][j] = wmma_bf16(af[i], bfr[j], acc[i][j]);

#pragma unroll
        for (int i = 0; i < 2; ++i)
#pragma unroll
            for (int j = 0; j < 4; ++j)
#pragma unroll
                for (int r = 0; r < 8; ++r)
                    Sm[32 * w + 16 * i + mbase + r][16 * j + ncol] = acc[i][j][r];
    }
    __syncthreads();

    // ---- softmax over row t with CPB bias + shift mask ----
    {
        float srow[WS_];
        float mx = -1e30f;
#pragma unroll 8
        for (int n = 0; n < WS_; ++n) {
            int   rel = t - n + (WS_ - 1);
            float bv  = 16.0f / (1.0f + __expf(-tbl[rel * H_ + head]));
            float s   = Sm[t][n] + bv;
            if (masked && ((t >> 5) != (n >> 5))) s -= 100.0f;
            srow[n] = s;
            mx = fmaxf(mx, s);
        }
        float sum = 0.f;
#pragma unroll 8
        for (int n = 0; n < WS_; ++n) { srow[n] = __expf(srow[n] - mx); sum += srow[n]; }
        float inv = 1.0f / sum;
#pragma unroll 8
        for (int n = 0; n < WS_; ++n) Pm[t][n] = f2bf(srow[n] * inv);
    }
    __syncthreads();

    // ---- O = P @ V : each wave 32x32 out, 2x2 tiles, K=64 (2 steps) ----
    {
        v8f o[2][2];
#pragma unroll
        for (int i = 0; i < 2; ++i)
#pragma unroll
            for (int j = 0; j < 2; ++j)
#pragma unroll
                for (int r = 0; r < 8; ++r) o[i][j][r] = 0.f;

#pragma unroll
        for (int ks = 0; ks < 2; ++ks) {
            v16bf af[2], bfr[2];
#pragma unroll
            for (int i = 0; i < 2; ++i)
                af[i] = load_frag(&Pm[0][0] + ks * 32, 32 * w + 16 * i, 72);
#pragma unroll
            for (int j = 0; j < 2; ++j)
                bfr[j] = load_frag(&Vt[0][0] + ks * 32, 16 * j, 72);
#pragma unroll
            for (int i = 0; i < 2; ++i)
#pragma unroll
                for (int j = 0; j < 2; ++j) o[i][j] = wmma_bf16(af[i], bfr[j], o[i][j]);
        }

#pragma unroll
        for (int i = 0; i < 2; ++i)
#pragma unroll
            for (int j = 0; j < 2; ++j)
#pragma unroll
                for (int r = 0; r < 8; ++r) {
                    int m = 32 * w + 16 * i + mbase + r;
                    int d = 16 * j + ncol;
                    attn_out[(size_t)(win * WS_ + m) * C_ + head * HD_ + d] =
                        f2bf(o[i][j][r]);
                }
    }
}

// ---------------------------------------------------------------------------
// out[row] = shortcut[row] + LN(y[src_row]);   src handles the +SHIFT un-roll.
// Optionally also writes a bf16 copy (feeds the next GEMM).
// ---------------------------------------------------------------------------
__global__ __launch_bounds__(256)
void ln_res_kernel(const float* __restrict__ y, const float* __restrict__ shortcut,
                   const float* __restrict__ w, const float* __restrict__ b,
                   float* __restrict__ out, bf16* __restrict__ outb, int shifted) {
    __shared__ float red[256];
    int row = blockIdx.x;
    int srow;
    if (shifted) {
        int bb = row >> 12, l = row & (L_ - 1);
        srow = (bb << 12) | ((l - SHIFT_) & (L_ - 1));
    } else srow = row;

    const float* yr = y + (size_t)srow * C_;
    int t = threadIdx.x;
    float v0 = yr[t], v1 = yr[t + 256];

    red[t] = v0 + v1;
    __syncthreads();
    for (int s = 128; s > 0; s >>= 1) { if (t < s) red[t] += red[t + s]; __syncthreads(); }
    float mean = red[0] * (1.0f / C_);
    __syncthreads();

    float d0 = v0 - mean, d1 = v1 - mean;
    red[t] = d0 * d0 + d1 * d1;
    __syncthreads();
    for (int s = 128; s > 0; s >>= 1) { if (t < s) red[t] += red[t + s]; __syncthreads(); }
    float rstd = rsqrtf(red[0] * (1.0f / C_) + 1e-5f);

    size_t base = (size_t)row * C_;
    float o0 = shortcut[base + t]       + (d0 * rstd) * w[t]       + b[t];
    float o1 = shortcut[base + t + 256] + (d1 * rstd) * w[t + 256] + b[t + 256];
    out[base + t]       = o0;
    out[base + t + 256] = o1;
    if (outb) { outb[base + t] = f2bf(o0); outb[base + t + 256] = f2bf(o1); }
}

// ---------------------------------------------------------------------------
// Launch
// ---------------------------------------------------------------------------
extern "C" void kernel_launch(void* const* d_in, const int* in_sizes, int n_in,
                              void* d_out, int out_size, void* d_ws, size_t ws_size,
                              hipStream_t stream) {
    const float* x       = (const float*)d_in[0];
    const float* qkv_w   = (const float*)d_in[1];
    const float* q_bias  = (const float*)d_in[2];
    const float* v_bias  = (const float*)d_in[3];
    const float* lscale  = (const float*)d_in[4];
    const float* cpb_w1  = (const float*)d_in[5];
    const float* cpb_b1  = (const float*)d_in[6];
    const float* cpb_w2  = (const float*)d_in[7];
    const float* proj_w  = (const float*)d_in[8];
    const float* proj_b  = (const float*)d_in[9];
    const float* norm1_w = (const float*)d_in[10];
    const float* norm1_b = (const float*)d_in[11];
    const float* norm2_w = (const float*)d_in[12];
    const float* norm2_b = (const float*)d_in[13];
    const float* fc1_w   = (const float*)d_in[14];
    const float* fc1_b   = (const float*)d_in[15];
    const float* fc2_w   = (const float*)d_in[16];
    const float* fc2_b   = (const float*)d_in[17];

    char* ws = (char*)d_ws;
    size_t off = 0;
    auto take = [&](size_t bytes) { size_t r = off; off += (bytes + 255) & ~(size_t)255; return r; };

    size_t o_xwbf   = take((size_t)MT_ * C_ * 2);
    size_t o_qkvwbf = take((size_t)3 * C_ * C_ * 2);
    size_t o_projwbf= take((size_t)C_ * C_ * 2);
    size_t o_fc1wbf = take((size_t)HID_ * C_ * 2);
    size_t o_fc2wbf = take((size_t)C_ * HID_ * 2);
    size_t o_qkvb   = take((size_t)3 * C_ * 4);
    size_t o_tbl    = take((size_t)127 * H_ * 4);
    size_t o_qkv    = take((size_t)MT_ * 3 * C_ * 4);   // 192MB; recycled below
    size_t o_attn   = take((size_t)MT_ * C_ * 2);
    size_t o_x1     = take((size_t)MT_ * C_ * 4);
    size_t o_x1bf   = take((size_t)MT_ * C_ * 2);

    bf16*  xw_bf   = (bf16*)(ws + o_xwbf);
    bf16*  qkvw_bf = (bf16*)(ws + o_qkvwbf);
    bf16*  projw_bf= (bf16*)(ws + o_projwbf);
    bf16*  fc1w_bf = (bf16*)(ws + o_fc1wbf);
    bf16*  fc2w_bf = (bf16*)(ws + o_fc2wbf);
    float* qkvb    = (float*)(ws + o_qkvb);
    float* tbl     = (float*)(ws + o_tbl);
    float* qkvf    = (float*)(ws + o_qkv);
    bf16*  attn_bf = (bf16*)(ws + o_attn);
    float* x1      = (float*)(ws + o_x1);
    bf16*  x1bf    = (bf16*)(ws + o_x1bf);
    // region reuse inside the 192MB qkv block (dead after attention):
    float* proj_out = (float*)(ws + o_qkv);                         // 64MB
    bf16*  h_bf     = (bf16*)(ws + o_qkv + (size_t)MT_ * C_ * 4);   // 128MB
    float* fc2_out  = (float*)(ws + o_qkv);                         // reuse proj_out

    // ---- prep ----
    cvt_kernel<<<512, 256, 0, stream>>>(qkv_w,  qkvw_bf, (long)3 * C_ * C_);
    cvt_kernel<<<256, 256, 0, stream>>>(proj_w, projw_bf, (long)C_ * C_);
    cvt_kernel<<<512, 256, 0, stream>>>(fc1_w,  fc1w_bf, (long)HID_ * C_);
    cvt_kernel<<<512, 256, 0, stream>>>(fc2_w,  fc2w_bf, (long)C_ * HID_);
    shiftcvt_kernel<<<4096, 256, 0, stream>>>(x, xw_bf);
    qkvb_kernel<<<6, 256, 0, stream>>>(q_bias, v_bias, qkvb);
    cpb_kernel<<<127, 128, 0, stream>>>(cpb_w1, cpb_b1, cpb_w2, tbl);

    // ---- qkv = xw @ qkv_w^T + qkv_b ----
    gemm_bf16<0><<<dim3(12, 256), 256, 0, stream>>>(xw_bf, qkvw_bf, qkvb,
                                                    qkvf, nullptr, MT_, 3 * C_, C_);
    // ---- windowed cosine attention ----
    attn_kernel<<<(MT_ / WS_) * H_, 64, 0, stream>>>(qkvf, lscale, tbl, attn_bf);
    // ---- proj ----
    gemm_bf16<0><<<dim3(4, 256), 256, 0, stream>>>(attn_bf, projw_bf, proj_b,
                                                   proj_out, nullptr, MT_, C_, C_);
    // ---- x1 = x + LN(unshift(proj_out)) ----
    ln_res_kernel<<<MT_, 256, 0, stream>>>(proj_out, x, norm1_w, norm1_b, x1, x1bf, 1);
    // ---- MLP ----
    gemm_bf16<1><<<dim3(16, 256), 256, 0, stream>>>(x1bf, fc1w_bf, fc1_b,
                                                    nullptr, h_bf, MT_, HID_, C_);
    gemm_bf16<0><<<dim3(4, 256), 256, 0, stream>>>(h_bf, fc2w_bf, fc2_b,
                                                   fc2_out, nullptr, MT_, C_, HID_);
    // ---- out = x1 + LN(fc2_out) ----
    ln_res_kernel<<<MT_, 256, 0, stream>>>(fc2_out, x1, norm2_w, norm2_b,
                                           (float*)d_out, nullptr, 0);
}
